// scaled_dot_product_attention_32392643346715
// MI455X (gfx1250) — compile-verified, matrix operand order checked by reading, not run on hardware
//
#include <hip/hip_runtime.h>

// Scaled-dot-product attention forward, returning (output, attention_weights).
// B=2, H=16, S=2048, Dh=64, fp32.  Matmuls on v_wmma_f32_16x16x32_f16 (f32
// inputs converted to f16, f32 accumulate).  Full 16x2048 f32 score tile in
// dynamic LDS (CDNA5 320KB/WGP) -> weights stream to HBM exactly once with
// non-temporal stores.  V tiles are double-buffered into LDS with
// global_load_async_to_lds_b128 (ASYNCcnt) during the PV phase.

typedef __attribute__((ext_vector_type(16))) _Float16 v16h;
typedef __attribute__((ext_vector_type(8)))  float    v8f;
typedef __attribute__((ext_vector_type(4)))  float    f32x4;

#define SDIM 2048
#define DH   64
#define ROWSTRIDE (SDIM + 20)   // 20 mod 64 -> 16 distinct row banks, 16B aligned

__device__ __forceinline__ void cvt8(v16h& d, int e0, const float* __restrict__ p) {
  const f32x4* p4 = (const f32x4*)p;
  f32x4 x = p4[0];
  f32x4 y = p4[1];
  d[e0 + 0] = (_Float16)x.x; d[e0 + 1] = (_Float16)x.y;
  d[e0 + 2] = (_Float16)x.z; d[e0 + 3] = (_Float16)x.w;
  d[e0 + 4] = (_Float16)y.x; d[e0 + 5] = (_Float16)y.y;
  d[e0 + 6] = (_Float16)y.z; d[e0 + 7] = (_Float16)y.w;
}

__device__ __forceinline__ void cvt4(v16h& d, int e0, f32x4 x) {
  d[e0 + 0] = (_Float16)x.x; d[e0 + 1] = (_Float16)x.y;
  d[e0 + 2] = (_Float16)x.z; d[e0 + 3] = (_Float16)x.w;
}

__device__ __forceinline__ v8f wmma_f16(v16h a, v16h b, v8f c) {
  return __builtin_amdgcn_wmma_f32_16x16x32_f16(false, a, false, b, (short)0, c,
                                                false, false);
}

// Async-copy one 32x64 f32 V tile (8KB) into LDS: 128 threads x 4 x b128.
__device__ __forceinline__ void stage_v_async(const float* __restrict__ gsrc,
                                              unsigned ldsbase, int t) {
#pragma unroll
  for (int i = 0; i < 4; ++i) {
    const int j = i * 128 + t;                       // float4 index in tile
    unsigned la = ldsbase + (unsigned)j * 16u;
    unsigned long long ga =
        (unsigned long long)gsrc + (unsigned long long)j * 16ull;
    asm volatile("global_load_async_to_lds_b128 %0, %1, off"
                 :: "v"(la), "v"(ga) : "memory");
  }
}

extern "C" __global__ __launch_bounds__(128)
void attn_fwd_kernel(const float* __restrict__ Q, const float* __restrict__ K,
                     const float* __restrict__ V, const float* __restrict__ M,
                     float* __restrict__ Out, float* __restrict__ W)
{
  extern __shared__ float smem[];
  float* sc    = smem;                          // [16][ROWSTRIDE] scores/exp
  float* redm  = sc + 16 * ROWSTRIDE;           // [4][16] row-max partials
  float* reds  = redm + 64;                     // [4][16] row-sum partials
  float* rinv  = reds + 64;                     // [16]    1/rowsum
  float* vbase = rinv + 16;                     // [2][32][64] V double buffer

  const int t    = threadIdx.x;
  const int lane = t & 31;
  const int wave = t >> 5;
  const int bh    = blockIdx.y;                 // b*H + h
  const int qbase = blockIdx.x * 16;

  const int col = lane & 15;                    // A-row / B-col of this lane
  const int dlo = (lane & 16) ? 8 : 0;          // K-dim sub-offset per half
  const int hi8 = (lane & 16) ? 8 : 0;          // C/D row offset per half

  const size_t bhS = (size_t)bh * SDIM;

  // ---- Q fragment: A-matrix 16x64 as two 16x32 f16 fragments -------------
  const float* qrow = Q + (bhS + qbase + col) * DH;
  v16h a0, a1;
  cvt8(a0, 0, qrow + dlo);       cvt8(a0, 8, qrow + 16 + dlo);
  cvt8(a1, 0, qrow + 32 + dlo);  cvt8(a1, 8, qrow + 48 + dlo);

  // ---- Phase 1: raw logits * 0.125 -> LDS (wave w: cols [w*512,w*512+512))
  for (int ct = wave * 32; ct < wave * 32 + 32; ++ct) {
    const int kb = ct * 16;
    const float* krow = K + (bhS + kb + col) * DH;
    v16h b0, b1;
    cvt8(b0, 0, krow + dlo);       cvt8(b0, 8, krow + 16 + dlo);
    cvt8(b1, 0, krow + 32 + dlo);  cvt8(b1, 8, krow + 48 + dlo);
    v8f acc = {};
    acc = wmma_f16(a0, b0, acc);
    acc = wmma_f16(a1, b1, acc);
#pragma unroll
    for (int e = 0; e < 8; ++e)
      sc[(e + hi8) * ROWSTRIDE + kb + col] = acc[e] * 0.125f;
  }
  __syncthreads();

  // ---- Phase 2: fused mask + softmax over LDS tile (float4 streams) ------
  const int srow = t & 15;                      // softmax row of this thread
  const int sseg = t >> 4;                      // 8 segments of 256 cols
  f32x4* seg4 = (f32x4*)(sc + srow * ROWSTRIDE + sseg * 256);
  const f32x4* mk4 = (const f32x4*)(M + (size_t)(qbase + srow) * SDIM + sseg * 256);

  float mx = -3.402823466e38f;
  for (int i = 0; i < 64; ++i) {
    f32x4 x = seg4[i];
    f32x4 mk = mk4[i];
    x.x = fmaf(mk.x, -1e9f, x.x);  x.y = fmaf(mk.y, -1e9f, x.y);
    x.z = fmaf(mk.z, -1e9f, x.z);  x.w = fmaf(mk.w, -1e9f, x.w);
    seg4[i] = x;
    mx = fmaxf(mx, fmaxf(fmaxf(x.x, x.y), fmaxf(x.z, x.w)));
  }
  mx = fmaxf(mx, __shfl_xor(mx, 16, 32));
  if (lane < 16) redm[wave * 16 + lane] = mx;
  __syncthreads();
  const float rm = fmaxf(fmaxf(redm[srow], redm[16 + srow]),
                         fmaxf(redm[32 + srow], redm[48 + srow]));

  float ssum = 0.f;
  for (int i = 0; i < 64; ++i) {
    f32x4 x = seg4[i];
    x.x = __expf(x.x - rm);  x.y = __expf(x.y - rm);
    x.z = __expf(x.z - rm);  x.w = __expf(x.w - rm);
    seg4[i] = x;                                // unnormalized exp back to LDS
    ssum += (x.x + x.y) + (x.z + x.w);
  }
  ssum += __shfl_xor(ssum, 16, 32);
  if (lane < 16) reds[wave * 16 + lane] = ssum;
  __syncthreads();
  const float rs = reds[srow] + reds[16 + srow] + reds[32 + srow] + reds[48 + srow];
  if (t < 16) rinv[t] = 1.0f / rs;              // thread t owns row t here
  __syncthreads();

  // Kick off async staging of V tile 0 so it overlaps the weight stream.
  const unsigned vb0 = (unsigned)(unsigned long long)vbase;
  stage_v_async(V + bhS * DH, vb0, t);

  // ---- Phase 3: normalized weights -> HBM, non-temporal float4 stores ----
  for (int i = 0; i < 64; ++i) {
    const int f4 = i * 128 + t;                 // 8192 float4s per block
    const int r  = f4 >> 9;                     // 512 float4 per row
    const int c4 = f4 & 511;
    f32x4 x = *((const f32x4*)(sc + r * ROWSTRIDE) + c4);
    const float iv = rinv[r];
    x.x *= iv; x.y *= iv; x.z *= iv; x.w *= iv;
    __builtin_nontemporal_store(
        x, (f32x4*)(W + (bhS + qbase + r) * (size_t)SDIM) + c4);
  }

  // ---- Phase 4: PV.  Wave w owns d-columns [w*16, w*16+16); all waves ----
  // share async-staged 32x64 V tiles, double buffered; full k contraction.
  const int dcol = wave * 16 + col;
  v8f o = {};
  for (int kt = 0; kt < 64; ++kt) {
    asm volatile("s_wait_asynccnt 0" ::: "memory");
    __syncthreads();                            // tile kt resident in buf kt&1
    if (kt + 1 < 64)
      stage_v_async(V + (bhS + (size_t)(kt + 1) * 32) * DH,
                    vb0 + (unsigned)(((kt + 1) & 1) * 8192), t);

    const int kb = kt * 32;
    const float* vt = vbase + (kt & 1) * 2048;  // [32][64] f32
    const float* arow = sc + col * ROWSTRIDE + kb + dlo;
    v16h ap, bv;
    cvt4(ap, 0,  *(const f32x4*)(arow));        // k = kb+dlo .. +3
    cvt4(ap, 4,  *(const f32x4*)(arow + 4));
    cvt4(ap, 8,  *(const f32x4*)(arow + 16));
    cvt4(ap, 12, *(const f32x4*)(arow + 20));
#pragma unroll
    for (int e = 0; e < 16; ++e) {
      const int km = ((e >> 3) << 4) + dlo + (e & 7);
      bv[e] = (_Float16)vt[km * DH + dcol];
    }
    o = wmma_f16(ap, bv, o);
  }

  // ---- Phase 5: scale by 1/rowsum, store output --------------------------
#pragma unroll
  for (int e = 0; e < 8; ++e) {
    const int r = e + hi8;
    Out[(bhS + qbase + r) * (size_t)DH + dcol] = o[e] * rinv[r];
  }
}

extern "C" void kernel_launch(void* const* d_in, const int* in_sizes, int n_in,
                              void* d_out, int out_size, void* d_ws, size_t ws_size,
                              hipStream_t stream) {
  (void)in_sizes; (void)n_in; (void)out_size; (void)d_ws; (void)ws_size;
  const float* Q = (const float*)d_in[0];
  const float* K = (const float*)d_in[1];
  const float* V = (const float*)d_in[2];
  const float* M = (const float*)d_in[3];

  const int B = 2, H = 16;
  float* Out = (float*)d_out;                                   // [B,H,S,Dh]
  float* W   = Out + (size_t)B * H * SDIM * DH;                 // [B,H,S,S]

  const int smem_bytes =
      (16 * ROWSTRIDE + 64 + 64 + 16 + 2 * 32 * 64) * (int)sizeof(float);
  hipFuncSetAttribute((const void*)attn_fwd_kernel,
                      hipFuncAttributeMaxDynamicSharedMemorySize, smem_bytes);

  dim3 grid(SDIM / 16, B * H);   // 128 query tiles x 32 heads
  dim3 block(128);               // 4 waves
  attn_fwd_kernel<<<grid, block, smem_bytes, stream>>>(Q, K, V, M, Out, W);
}
